// KahlerAttention_51324859187419
// MI455X (gfx1250) — matrix operand order repaired and checked by reading, not compiled, not run on hardware
//
#include <hip/hip_runtime.h>
#include <stdint.h>

typedef unsigned short u16;
typedef __attribute__((ext_vector_type(16))) __bf16 v16bf;
typedef __attribute__((ext_vector_type(8)))  float  v8f;
typedef __attribute__((ext_vector_type(4)))  unsigned tdm_u4;
typedef __attribute__((ext_vector_type(8)))  int      tdm_i8;
typedef __attribute__((ext_vector_type(4)))  int      tdm_i4;

#if __has_builtin(__builtin_amdgcn_tensor_load_to_lds)
#define HAVE_TDM 1
#else
#define HAVE_TDM 0
#endif

// ---------------------------------------------------------------- helpers

__device__ __forceinline__ u16 f32_to_bf16(float f) {
  unsigned u = __float_as_uint(f);
  u += 0x7fffu + ((u >> 16) & 1u);   // round-to-nearest-even
  return (u16)(u >> 16);
}

union FragU { v16bf v; uint4 u[2]; };

// A fragment (16x32 bf16) from LDS tile stored [row][k], row stride ldw (u16 elems).
// ISA layout: lane m(+16*half); VGPR0-3 = K[half*8 .. half*8+7], VGPR4-7 = K[16+half*8 ..]
__device__ __forceinline__ v16bf load_a_frag(const u16* lds, int ldw, int tile_m,
                                             int k0, int lane) {
  int m = lane & 15, lh = lane >> 4;
  const u16* p = lds + (size_t)(tile_m + m) * ldw + k0 + lh * 8;
  FragU f;
  f.u[0] = *(const uint4*)(p);
  f.u[1] = *(const uint4*)(p + 16);
  return f.v;
}

// B fragment (32x16 bf16) from LDS tile stored transposed [n][k].
// ISA layout: lane n(+16*half); VGPR0-3 = K[half*8 .. half*8+7], VGPR4-7 = K[16+half*8 ..]
__device__ __forceinline__ v16bf load_b_frag(const u16* lds, int ldw, int tile_n,
                                             int k0, int lane) {
  int n = lane & 15, lh = lane >> 4;
  const u16* p = lds + (size_t)(tile_n + n) * ldw + k0 + lh * 8;
  FragU f;
  f.u[0] = *(const uint4*)(p);
  f.u[1] = *(const uint4*)(p + 16);
  return f.v;
}

__device__ __forceinline__ v8f wmma_bf16(v16bf a, v16bf b, v8f c) {
  return __builtin_amdgcn_wmma_f32_16x16x32_bf16(false, a, false, b,
                                                 (short)0, c, false, false);
}

// Async DMA: global -> LDS, 16 bytes per lane, tracked by ASYNCcnt.
// LDS operand wants the 32-bit LDS byte offset; on amdgcn the generic LDS
// address carries the LDS offset in its low 32 bits (ISA flat->LDS truncation).
__device__ __forceinline__ void async_b128(void* lds_ptr, const void* gptr) {
  unsigned loff = (unsigned)(uintptr_t)lds_ptr;
  asm volatile("global_load_async_to_lds_b128 %0, %1, off"
               :: "v"(loff), "v"(gptr) : "memory");
}
__device__ __forceinline__ void wait_async_0() {
  asm volatile("s_wait_asynccnt 0x0" ::: "memory");
}
__device__ __forceinline__ void wait_async_1() {
  asm volatile("s_wait_asynccnt 0x1" ::: "memory");
}

// TDM: one descriptor-driven DMA of a 2D bf16 tile (tile_k x tile_n) from a
// row-major [1024][1024] tensor into LDS with hardware row padding
// (64B data rows -> +16B pad, matching the GLDW=40 u16 LDS stride).
// D# packing per CDNA5 ISA §8.3 (group0) / §8.4 (group1).
__device__ __forceinline__ void tdm_load_2d(unsigned lds_off, const void* gaddr,
                                            unsigned tile_k, unsigned tile_n) {
#if HAVE_TDM
  uint64_t ga = (uint64_t)(uintptr_t)gaddr;
  tdm_u4 g0;
  g0[0] = 1u;                                        // count=1 (valid), user mode
  g0[1] = lds_off;                                   // lds_addr (bytes)
  g0[2] = (unsigned)(ga & 0xffffffffu);              // global_addr[31:0]
  g0[3] = (unsigned)((ga >> 32) & 0x01ffffffu)       // global_addr[56:32]
        | (2u << 30);                                // type=2 ("image")
  tdm_i8 g1;
  g1[0] = (int)((1u << 16)                           // data_size: 2 bytes
              | (1u << 20)                           // pad_enable
              | (3u << 22)                           // pad_interval: 16 DWORDs
              | (3u << 25));                         // pad_amount: 4 DWORDs
  g1[1] = (int)(1024u << 16);                        // tensor_dim0[15:0]=1024
  g1[2] = (int)(1024u << 16);                        // dim0[31:16]=0, dim1[15:0]=1024
  g1[3] = (int)(tile_k << 16);                       // dim1[31:16]=0, tile_dim0
  g1[4] = (int)tile_n;                               // tile_dim1 (tile_dim2=0)
  g1[5] = 1024;                                      // tensor_dim0_stride[31:0]
  g1[6] = 0;                                         // stride[47:32], dim1_stride lo
  g1[7] = 0;
  tdm_i4 gz = {0, 0, 0, 0};
#if defined(__clang_major__) && (__clang_major__ >= 23)
  tdm_i8 gz8 = {0, 0, 0, 0, 0, 0, 0, 0};
  __builtin_amdgcn_tensor_load_to_lds(g0, g1, gz, gz, gz8, 0);
#else
  __builtin_amdgcn_tensor_load_to_lds(g0, g1, gz, gz, 0);
#endif
#else
  (void)lds_off; (void)gaddr; (void)tile_k; (void)tile_n;
#endif
}

// ---------------------------------------------------------------- small kernels

__global__ __launch_bounds__(256) void cvt_bf16_kernel(const float* __restrict__ in,
                                                       u16* __restrict__ out, int n) {
  int i = blockIdx.x * 256 + threadIdx.x;
  if (i < n) out[i] = f32_to_bf16(in[i]);
}

// Fold quaternion block structure into dense transposed weight Wt[n=out][k=in] (bf16)
// and folded bias (fp32).  out[e*256+o] = sum_{c,d} x[c*256+d] * sgn[e][c]*W[widx[e][c]][d][o]
__global__ __launch_bounds__(256) void build_weights_kernel(const float* __restrict__ W,
                                                            const float* __restrict__ bv,
                                                            u16* __restrict__ Wt,
                                                            float* __restrict__ biasf) {
  const int   widx[4][4] = {{0,1,2,3},{1,0,3,2},{2,3,0,1},{3,2,1,0}};
  const float sgn [4][4] = {{1.f,-1.f,-1.f,-1.f},{1.f,1.f,-1.f,1.f},
                            {1.f,1.f,1.f,-1.f},{1.f,-1.f,1.f,1.f}};
  int idx = blockIdx.x * 256 + threadIdx.x;        // 0 .. 1024*1024-1
  int row = idx >> 10;                             // out index e*256+o
  int col = idx & 1023;                            // in  index c*256+d
  int e = row >> 8, o = row & 255;
  int c = col >> 8, d = col & 255;
  int w = widx[e][c];
  Wt[idx] = f32_to_bf16(sgn[e][c] * W[(w << 16) + (d << 8) + o]);
  if (idx < 1024) {
    int ee = idx >> 8, oo = idx & 255;
    float acc = 0.f;
    for (int cc = 0; cc < 4; ++cc)
      acc += sgn[ee][cc] * bv[(widx[ee][cc] << 8) + oo];
    biasf[idx] = acc;
  }
}

// scalar score bias = mean(sin(J[:n])) * 0.1
__global__ __launch_bounds__(256) void score_bias_kernel(const float* __restrict__ J,
                                                         int n, float* __restrict__ out) {
  __shared__ float red[256];
  float s = 0.f;
  for (int i = threadIdx.x; i < n; i += 256) s += sinf(J[i]);
  red[threadIdx.x] = s;
  __syncthreads();
  for (int st = 128; st > 0; st >>= 1) {
    if (threadIdx.x < (unsigned)st) red[threadIdx.x] += red[threadIdx.x + st];
    __syncthreads();
  }
  if (threadIdx.x == 0) out[0] = red[0] / (float)n * 0.1f;
}

// quaternion normalization: components live at o, o+256, o+512, o+768 of each row
__global__ __launch_bounds__(256) void quat_norm_kernel(const float* __restrict__ Y,
                                                        u16* __restrict__ out) {
  int row = blockIdx.x, o = threadIdx.x;
  const float* yr = Y + (size_t)row * 1024;
  float r = yr[o], i = yr[256 + o], j = yr[512 + o], k = yr[768 + o];
  float inv = rsqrtf(r * r + i * i + j * j + k * k + 1e-6f);
  u16* orow = out + (size_t)row * 1024;
  orow[o]       = f32_to_bf16(r * inv);
  orow[256 + o] = f32_to_bf16(i * inv);
  orow[512 + o] = f32_to_bf16(j * inv);
  orow[768 + o] = f32_to_bf16(k * inv);
}

// ---------------------------------------------------------------- GEMM (WMMA bf16)
// C[M][1024] = A[M][1024] @ Bt[1024][1024]^T + bias   (Bt stored [n][k])
// Block tile 64x256, 8 waves (2 x 4), wave tile 32x64 (2x4 WMMA tiles), K-step 32.
// Double-buffered LDS: A slab via async global->LDS DMA (ASYNCcnt), B slab via
// a single TDM tensor_load_to_lds per slab (TENSORcnt), both pipelined.
#define GLDW 40   // padded LDS row stride in u16 (80B: what the TDM pad produces)

__global__ __launch_bounds__(256) void gemm_bf16_kernel(const u16* __restrict__ A,
                                                        const u16* __restrict__ Bt,
                                                        const float* __restrict__ bias,
                                                        void* __restrict__ out,
                                                        int M, int outMode) {
  const int K = 1024, N = 1024;
  __shared__ __align__(16) u16 sA[2][64 * GLDW];
  __shared__ __align__(16) u16 sB[2][256 * GLDW];
  int tid = threadIdx.x, lane = tid & 31, w = tid >> 5;
  int m0 = blockIdx.y * 64, n0 = blockIdx.x * 256;
  int wm = (w & 1) * 32, wn = (w >> 1) * 64;
  int arow = tid >> 2, aq = tid & 3;   // A tile: 256 chunks of 8 u16, one per thread

  auto issue_slab = [&](int buf, int k0) {
    // A slab: 1 async op per thread
    async_b128((void*)(sA[buf] + arow * GLDW + aq * 8),
               (const void*)(A + (size_t)(m0 + arow) * K + k0 + aq * 8));
#if HAVE_TDM
    // B slab: one TDM descriptor, issued by wave 0 only (EXEC-independent DMA)
    if (w == 0)
      tdm_load_2d((unsigned)(uintptr_t)(void*)sB[buf],
                  (const void*)(Bt + (size_t)n0 * K + k0), 32u, 256u);
#else
    for (int i = 0; i < 4; ++i) {
      int c = tid + i * 256;           // 1024 chunks for 256 rows x 32 k
      int row = c >> 2, qq = c & 3;
      async_b128((void*)(sB[buf] + row * GLDW + qq * 8),
                 (const void*)(Bt + (size_t)(n0 + row) * K + k0 + qq * 8));
    }
#endif
  };

  v8f acc[2][4] = {};
  issue_slab(0, 0);
  for (int kt = 0; kt < K / 32; ++kt) {
    int buf = kt & 1;
    bool more = (kt + 1 < K / 32);
    if (more) issue_slab(buf ^ 1, (kt + 1) * 32);
    // wait only for the current slab's DMAs; next slab's stay in flight
#if HAVE_TDM
    if (more) wait_async_1(); else wait_async_0();
    if (w == 0) {
      if (more) __builtin_amdgcn_s_wait_tensorcnt(1);
      else      __builtin_amdgcn_s_wait_tensorcnt(0);
    }
#else
    if (more) asm volatile("s_wait_asynccnt 0x5" ::: "memory");
    else      wait_async_0();
#endif
    __syncthreads();

    v16bf af[2], bfr[4];
    af[0] = load_a_frag(sA[buf], GLDW, wm + 0,  0, lane);
    af[1] = load_a_frag(sA[buf], GLDW, wm + 16, 0, lane);
    for (int nt = 0; nt < 4; ++nt)
      bfr[nt] = load_b_frag(sB[buf], GLDW, wn + nt * 16, 0, lane);
    for (int mt = 0; mt < 2; ++mt)
      for (int nt = 0; nt < 4; ++nt)
        acc[mt][nt] = wmma_bf16(af[mt], bfr[nt], acc[mt][nt]);
    __syncthreads();   // all waves done reading before slab is overwritten
  }

  // epilogue: C layout — element (m,n): vgpr m%8, lane n + 16*(m/8)
  int lane_n = lane & 15, lh = lane >> 4;
  for (int mt = 0; mt < 2; ++mt)
    for (int nt = 0; nt < 4; ++nt) {
      int bm = m0 + wm + mt * 16 + lh * 8;
      int bn = n0 + wn + nt * 16 + lane_n;
      float bv = bias[bn];
      for (int r = 0; r < 8; ++r) {
        float vv = acc[mt][nt][r] + bv;
        if (outMode == 0) ((float*)out)[(size_t)(bm + r) * N + bn] = vv;
        else              ((u16*)out)[(size_t)(bm + r) * N + bn] = f32_to_bf16(vv);
      }
    }
}

// ---------------------------------------------------------------- flash attention
// 4 waves / block; each wave owns 16 query rows of a 64-row Q tile; key tiles of 64.
#define FLDW 72   // padded LDS row stride (144B)

__global__ __launch_bounds__(128) void flash_attn_kernel(const u16* __restrict__ q,
                                                         const u16* __restrict__ k,
                                                         const u16* __restrict__ v,
                                                         u16* __restrict__ o,
                                                         const float* __restrict__ sbias_p,
                                                         int B, int T, int H) {
  const int D = 1024, Dh = 64;
  __shared__ __align__(16) u16 sQ [64 * FLDW];
  __shared__ __align__(16) u16 sK [64 * FLDW];
  __shared__ __align__(16) u16 sVT[64 * FLDW];   // V transposed: [out-dim][key-row]
  __shared__ __align__(16) u16 sP [64 * FLDW];   // probabilities, A-operand staging
  int tid = threadIdx.x, lane = tid & 31, w = tid >> 5;
  int bh = blockIdx.y;
  int b = bh / H, h = bh % H;
  int q0 = blockIdx.x * 64;
  const float scale = 0.125f;                    // 64^-0.5
  float sb = sbias_p[0];
  int lane_n = lane & 15, lh = lane >> 4;

  // stage Q tile once (64 rows x 64 cols bf16) via async DMA
  size_t baseQ = ((size_t)(b * T + q0)) * D + h * Dh;
  for (int i = 0; i < 4; ++i) {
    int c = tid + i * 128;           // 512 chunks of 8 u16
    int row = c >> 3, qq = c & 7;
    async_b128((void*)(sQ + row * FLDW + qq * 8),
               (const void*)(q + baseQ + (size_t)row * D + qq * 8));
  }

  float mrun[8], lrun[8];
  for (int r = 0; r < 8; ++r) { mrun[r] = -1e30f; lrun[r] = 0.f; }
  v8f accO[4] = {};

  for (int kt = 0; kt < (T >> 6); ++kt) {
    int kb = kt * 64;
    size_t baseK = ((size_t)(b * T + kb)) * D + h * Dh;
    __syncthreads();   // previous iteration done reading sK/sVT/sP
    for (int i = 0; i < 4; ++i) {
      int c = tid + i * 128;
      int row = c >> 3, qq = c & 7;
      async_b128((void*)(sK + row * FLDW + qq * 8),
                 (const void*)(k + baseK + (size_t)row * D + qq * 8));
    }
    for (int i = 0; i < 32; ++i) {   // V transpose, element-wise
      int idx = tid + i * 128;
      int kr = idx >> 6, d = idx & 63;
      sVT[d * FLDW + kr] = v[baseK + (size_t)kr * D + d];
    }
    wait_async_0();                  // Q (first iter) + K DMAs landed
    __syncthreads();

    // S strip (16 x 64) = Q(16xDh) . K^T : sK rows are exactly B^T layout [n][k]
    v8f accS[4] = {};
    for (int dk = 0; dk < Dh; dk += 32) {
      v16bf aq = load_a_frag(sQ, FLDW, w * 16, dk, lane);
      for (int nt = 0; nt < 4; ++nt) {
        v16bf bk = load_b_frag(sK, FLDW, nt * 16, dk, lane);
        accS[nt] = wmma_bf16(aq, bk, accS[nt]);
      }
    }

    // online softmax; row (w*16 + lh*8 + r) lives in vgpr r across one 16-lane half
    float pr[4][8];
    for (int r = 0; r < 8; ++r) {
      float rowmax = -1e30f;
      for (int nt = 0; nt < 4; ++nt) {
        float s = accS[nt][r] * scale + sb;
        pr[nt][r] = s;
        rowmax = fmaxf(rowmax, s);
      }
      for (int msk = 1; msk <= 8; msk <<= 1)
        rowmax = fmaxf(rowmax, __shfl_xor(rowmax, msk, 32));
      float newm = fmaxf(mrun[r], rowmax);
      float alpha = __expf(mrun[r] - newm);
      mrun[r] = newm;
      float rs = 0.f;
      for (int nt = 0; nt < 4; ++nt) {
        float p = __expf(pr[nt][r] - newm);
        pr[nt][r] = p;
        rs += p;
      }
      for (int msk = 1; msk <= 8; msk <<= 1)
        rs += __shfl_xor(rs, msk, 32);
      lrun[r] = lrun[r] * alpha + rs;
      for (int nt = 0; nt < 4; ++nt) accO[nt][r] *= alpha;
    }

    // stage P (bf16) in A-operand memory order [row][col]
    for (int nt = 0; nt < 4; ++nt)
      for (int r = 0; r < 8; ++r)
        sP[(size_t)(w * 16 + lh * 8 + r) * FLDW + nt * 16 + lane_n] =
            f32_to_bf16(pr[nt][r]);
    __syncthreads();

    // O += P(16x64) . V(64x64) ; sVT rows are B^T layout [n][k]
    for (int dk = 0; dk < 64; dk += 32) {
      v16bf ap = load_a_frag(sP, FLDW, w * 16, dk, lane);
      for (int nt = 0; nt < 4; ++nt) {
        v16bf bvf = load_b_frag(sVT, FLDW, nt * 16, dk, lane);
        accO[nt] = wmma_bf16(ap, bvf, accO[nt]);
      }
    }
  }

  // epilogue: normalize by row sum, write bf16
  for (int nt = 0; nt < 4; ++nt)
    for (int r = 0; r < 8; ++r) {
      int row = q0 + w * 16 + lh * 8 + r;
      int col = h * Dh + nt * 16 + lane_n;
      float val = accO[nt][r] / lrun[r];
      o[((size_t)(b * T + row)) * D + col] = f32_to_bf16(val);
    }
}

// ---------------------------------------------------------------- launcher

extern "C" void kernel_launch(void* const* d_in, const int* in_sizes, int n_in,
                              void* d_out, int out_size, void* d_ws, size_t ws_size,
                              hipStream_t stream) {
  (void)n_in; (void)out_size; (void)ws_size;
  const float* x = (const float*)d_in[0];
  const float* pW[4] = {(const float*)d_in[1], (const float*)d_in[3],
                        (const float*)d_in[5], (const float*)d_in[7]};
  const float* pb[4] = {(const float*)d_in[2], (const float*)d_in[4],
                        (const float*)d_in[6], (const float*)d_in[8]};
  const float* J = (const float*)d_in[9];

  const int B = 2, T = 2048, D = 1024, H = 16;   // reference shapes
  const int M = B * T;                            // 4096 tokens

  // workspace carve-out (~64 MB total)
  char* base = (char*)d_ws;
  size_t off = 0;
  auto walloc = [&](size_t bytes) -> void* {
    void* p = base + off;
    off = (off + bytes + 255) & ~(size_t)255;
    return p;
  };
  u16* xb = (u16*)walloc((size_t)M * D * 2);
  u16* Wt[4]; float* bfold[4];
  for (int p = 0; p < 4; ++p) {
    Wt[p]    = (u16*)walloc((size_t)D * D * 2);
    bfold[p] = (float*)walloc((size_t)D * 4);
  }
  float* sbias = (float*)walloc(256);
  float* stage = (float*)walloc((size_t)M * D * 4);
  u16* qn = (u16*)walloc((size_t)M * D * 2);
  u16* kn = (u16*)walloc((size_t)M * D * 2);
  u16* vn = (u16*)walloc((size_t)M * D * 2);
  u16* ao = (u16*)walloc((size_t)M * D * 2);

  // 1) x -> bf16
  cvt_bf16_kernel<<<(M * D) / 256, 256, 0, stream>>>(x, xb, M * D);

  // 2) fold quaternion structure into dense transposed bf16 weights + biases
  for (int p = 0; p < 4; ++p)
    build_weights_kernel<<<(D * D) / 256, 256, 0, stream>>>(pW[p], pb[p], Wt[p], bfold[p]);

  // 3) scalar attention bias = mean(sin(J[:min(T,len(J))])) * 0.1
  int nJ = in_sizes[9] < T ? in_sizes[9] : T;
  score_bias_kernel<<<1, 256, 0, stream>>>(J, nJ, sbias);

  dim3 ggrid(D / 256, M / 64);

  // 4) q projection -> fp32 stage -> quat-norm -> bf16
  gemm_bf16_kernel<<<ggrid, 256, 0, stream>>>(xb, Wt[0], bfold[0], stage, M, 0);
  quat_norm_kernel<<<M, 256, 0, stream>>>(stage, qn);

  // 5) k projection -> fp32 stage (reused) -> quat-norm -> bf16
  gemm_bf16_kernel<<<ggrid, 256, 0, stream>>>(xb, Wt[1], bfold[1], stage, M, 0);
  quat_norm_kernel<<<M, 256, 0, stream>>>(stage, kn);

  // 6) v projection -> bf16 directly (no norm)
  gemm_bf16_kernel<<<ggrid, 256, 0, stream>>>(xb, Wt[2], bfold[2], vn, M, 1);

  // 7) flash attention -> bf16 attention output
  flash_attn_kernel<<<dim3(T / 64, B * H), 128, 0, stream>>>(qn, kn, vn, ao, sbias, B, T, H);

  // 8) output projection -> fp32 d_out
  gemm_bf16_kernel<<<ggrid, 256, 0, stream>>>(ao, Wt[3], bfold[3], d_out, M, 0);
}
